// SpatialTemporalBlock_27676769255591
// MI455X (gfx1250) — matrix operand order repaired and checked by reading, not compiled, not run on hardware
//
#include <hip/hip_runtime.h>
#include <hip/hip_bf16.h>

#define NN 16384
#define CC 128
#define EE 524288
#define ET (EE + NN)   // edges including self-loops
#define SPLITS 2       // flash-attention key splits (flash-decoding style)

typedef __attribute__((ext_vector_type(16))) __bf16 v16bf;
typedef __attribute__((ext_vector_type(8)))  __bf16 v8bf;
typedef __attribute__((ext_vector_type(2)))  __bf16 v2bf;
typedef __attribute__((ext_vector_type(8)))  float  v8f;
typedef __attribute__((ext_vector_type(2)))  float  v2f;
typedef __attribute__((ext_vector_type(4)))  unsigned int v4u;
typedef __attribute__((ext_vector_type(8)))  int    v8i;
typedef __attribute__((ext_vector_type(4)))  int    v4i;

#if __has_builtin(__builtin_amdgcn_tensor_load_to_lds)
#define USE_TDM 1
#endif

__device__ __forceinline__ unsigned flipf(float f) {
  unsigned u = __float_as_uint(f);
  return (u & 0x80000000u) ? ~u : (u | 0x80000000u);
}
__device__ __forceinline__ float unflipf(unsigned m) {
  unsigned u = (m & 0x80000000u) ? (m & 0x7FFFFFFFu) : ~m;
  return __uint_as_float(u);
}

// ---------------- fp32 GEMM on matrix pipes: out[N,C] = X[N,C] @ W[C,C] ----------------
__global__ __launch_bounds__(128, 1)
void node_gemm_wmma(const float* __restrict__ X, const float* __restrict__ W,
                    float* __restrict__ out) {
  __shared__ float sX[16 * 128];
  const int tid = threadIdx.x;
  const int wave = tid >> 5;
  const int lane = tid & 31;
  const int m16 = lane & 15;
  const int khalf = lane >> 4;
  const int row0 = blockIdx.x * 16;

  for (int t = tid; t < 512; t += 128)
    ((float4*)sX)[t] = ((const float4*)(X + (size_t)row0 * CC))[t];
  __syncthreads();

  const int col0 = wave * 32;
  v8f c0 = {0.f, 0.f, 0.f, 0.f, 0.f, 0.f, 0.f, 0.f};
  v8f c1 = {0.f, 0.f, 0.f, 0.f, 0.f, 0.f, 0.f, 0.f};
#pragma unroll
  for (int k4 = 0; k4 < 32; ++k4) {
    const int k = k4 * 4 + khalf * 2;
    v2f a = *(const v2f*)(&sX[m16 * 128 + k]);
    v2f b0, b1;
    b0[0] = W[(size_t)k * CC + col0 + m16];
    b0[1] = W[(size_t)(k + 1) * CC + col0 + m16];
    b1[0] = W[(size_t)k * CC + col0 + 16 + m16];
    b1[1] = W[(size_t)(k + 1) * CC + col0 + 16 + m16];
    c0 = __builtin_amdgcn_wmma_f32_16x16x4_f32(false, a, false, b0, (short)0, c0, false, false);
    c1 = __builtin_amdgcn_wmma_f32_16x16x4_f32(false, a, false, b1, (short)0, c1, false, false);
  }
#pragma unroll
  for (int v = 0; v < 8; ++v) {
    const int row = row0 + v + khalf * 8;
    out[(size_t)row * CC + col0 + m16] = c0[v];
    out[(size_t)row * CC + col0 + 16 + m16] = c1[v];
  }
}

// as_out[r] = xw[r,:]·avs ; ad_out[r] = xw[r,:]·avd  (one wave per row)
__global__ void alpha_kernel(const float* __restrict__ xw, const float* __restrict__ avs,
                             const float* __restrict__ avd, float* __restrict__ as_out,
                             float* __restrict__ ad_out) {
  const int row = blockIdx.x * 8 + (threadIdx.x >> 5);
  const int lane = threadIdx.x & 31;
  const float4 v = *(const float4*)(xw + (size_t)row * CC + lane * 4);
  const float4 s = *(const float4*)(avs + lane * 4);
  const float4 d = *(const float4*)(avd + lane * 4);
  float ps = v.x * s.x + v.y * s.y + v.z * s.z + v.w * s.w;
  float pd = v.x * d.x + v.y * d.y + v.z * d.z + v.w * d.w;
#pragma unroll
  for (int off = 16; off >= 1; off >>= 1) {
    ps += __shfl_xor(ps, off);
    pd += __shfl_xor(pd, off);
  }
  if (lane == 0) { as_out[row] = ps; ad_out[row] = pd; }
}

__global__ void gat_init(unsigned* __restrict__ emax, float* __restrict__ denom,
                         float* __restrict__ h, const float* __restrict__ bias) {
  const int i = blockIdx.x * blockDim.x + threadIdx.x;
  if (i < NN * CC) h[i] = bias[i & (CC - 1)];
  if (i < NN) { emax[i] = 0x007FFFFFu; denom[i] = 0.f; }  // flip(-inf)
}

__global__ void edge_logits(const int* __restrict__ ei, const float* __restrict__ as_,
                            const float* __restrict__ ad_, float* __restrict__ ev,
                            unsigned* __restrict__ em) {
  const int i = blockIdx.x * blockDim.x + threadIdx.x;
  if (i >= ET) return;
  int s, d;
  if (i < EE) { s = ei[i]; d = ei[EE + i]; } else { s = d = i - EE; }
  float e = as_[s] + ad_[d];
  e = e > 0.f ? e : 0.2f * e;  // leaky_relu(0.2)
  ev[i] = e;
  atomicMax(&em[d], flipf(e));
}

__global__ void edge_exp(const int* __restrict__ ei, const float* __restrict__ ev,
                         const unsigned* __restrict__ em, float* __restrict__ pv,
                         float* __restrict__ denom) {
  const int i = blockIdx.x * blockDim.x + threadIdx.x;
  if (i >= ET) return;
  const int d = (i < EE) ? ei[EE + i] : (i - EE);
  const float p = __expf(ev[i] - unflipf(em[d]));
  pv[i] = p;
  atomicAdd(&denom[d], p);
}

__global__ void edge_aggregate(const int* __restrict__ ei, const float* __restrict__ pv,
                               const float* __restrict__ denom, const float* __restrict__ xw,
                               float* __restrict__ h) {
  const long long g = (long long)blockIdx.x * blockDim.x + threadIdx.x;
  if (g >= (long long)ET * 32) return;
  const int e = (int)(g >> 5);
  const int part = ((int)g & 31) * 4;
  int s, d;
  if (e < EE) { s = ei[e]; d = ei[EE + e]; } else { s = d = e - EE; }
  const float coeff = pv[e] / denom[d];
  const float4 v = *(const float4*)(xw + (size_t)s * CC + part);
  float* hp = h + (size_t)d * CC + part;
  atomicAdd(hp + 0, coeff * v.x);
  atomicAdd(hp + 1, coeff * v.y);
  atomicAdd(hp + 2, coeff * v.z);
  atomicAdd(hp + 3, coeff * v.w);
}

__global__ void relu_k(float* __restrict__ h, int n) {
  const int i = blockIdx.x * blockDim.x + threadIdx.x;
  if (i < n) h[i] = fmaxf(h[i], 0.f);
}

__global__ void cast_bf16_k(const float* __restrict__ in, __bf16* __restrict__ out, int n) {
  const int i = blockIdx.x * blockDim.x + threadIdx.x;
  if (i < n) out[i] = (__bf16)in[i];
}

// ---------------- Flash attention, transposed-S, key-split ----------------
// Block: 128 threads = 4 waves, 64 query rows; grid (NN/64, SPLITS).
// Each split covers NN/SPLITS keys and writes unnormalized partial O + (m, l).
__global__ __launch_bounds__(128, 1)
void flash_attn(const __bf16* __restrict__ qb, const __bf16* __restrict__ kvb,
                float* __restrict__ po, float* __restrict__ pm, float* __restrict__ pl) {
  __shared__ __bf16 sK[32 * 128];    // K tile, row-major [key][feat]
  __shared__ __bf16 sVT[128 * 32];   // V tile, transposed [feat][key]
  __shared__ __bf16 sP[4][16 * 32];  // per-wave P^T stored as [query][key]
  const int tid = threadIdx.x;
  const int wave = tid >> 5;
  const int lane = tid & 31;
  const int m16 = lane & 15;
  const int khalf = lane >> 4;
  const int split = blockIdx.y;
  const int qrow0 = blockIdx.x * 64 + wave * 16;
  const int key0 = split * (NN / SPLITS);
  const int NIT = (NN / SPLITS) / 32;

  // Q^T B-operand: B(k=feat,n=query): n=lane&15, 16 contiguous feats per lane-half
  v16bf bQ[4];
#pragma unroll
  for (int b = 0; b < 4; ++b)
    bQ[b] = *(const v16bf*)(qb + (size_t)(qrow0 + m16) * CC + b * 32 + khalf * 16);

  v8f oc[8];  // O^T tiles per feature chunk: M=fcol, N=query
#pragma unroll
  for (int f = 0; f < 8; ++f) oc[f] = (v8f){0.f, 0.f, 0.f, 0.f, 0.f, 0.f, 0.f, 0.f};
  float mstate = -3.0e38f, lstate = 0.f;  // per-lane = per-query (replicated x2)

  const float scale = 0.088388347648318447f;  // 1/sqrt(128)

  for (int kt = 0; kt < NIT; ++kt) {
    const int keybase = key0 + kt * 32;
    __syncthreads();
#if USE_TDM
    // TDM DMA fills sK (32x128 bf16 row-major) while threads build the transposed copy.
    if (wave == 0) {
      const unsigned lds_addr = (unsigned)(uintptr_t)(void*)&sK[0];
      const unsigned long long ga =
          (unsigned long long)(uintptr_t)(kvb + (size_t)keybase * CC);
      v4u g0;
      g0[0] = 1u;                                   // count=1, user descriptor
      g0[1] = lds_addr;                             // lds_addr [63:32]
      g0[2] = (unsigned)(ga & 0xFFFFFFFFu);         // global_addr [95:64]
      g0[3] = (unsigned)((ga >> 32) & 0x01FFFFFFu)  // global_addr [120:96]
              | (2u << 30);                         // type=2 ("image")
      v8i g1;
      g1[0] = (int)(1u << 16);       // workgroup_mask=0, data_size=1 (2B)
      g1[1] = (int)(128u << 16);     // tensor_dim0 = 128  (bits 79:48 lo half)
      g1[2] = (int)(32u << 16);      // tensor_dim0 hi=0 | tensor_dim1 = 32
      g1[3] = (int)(128u << 16);     // tensor_dim1 hi=0 | tile_dim0 = 128
      g1[4] = 32;                    // tile_dim1 = 32, tile_dim2 = 0
      g1[5] = 128;                   // tensor_dim0_stride = 128 (lo32)
      g1[6] = 0;
      g1[7] = 0;
      v4i z4 = {0, 0, 0, 0};
      v8i z8 = {0, 0, 0, 0, 0, 0, 0, 0};
      __builtin_amdgcn_tensor_load_to_lds(g0, g1, z4, z4, z8, 0);
      __builtin_amdgcn_s_wait_tensorcnt(0);
    }
    for (int t = tid; t < 256; t += 128) {
      const int r2 = (t >> 4) << 1;
      const int c8 = (t & 15) << 3;
      v8bf va = *(const v8bf*)(kvb + (size_t)(keybase + r2) * CC + c8);
      v8bf vb = *(const v8bf*)(kvb + (size_t)(keybase + r2 + 1) * CC + c8);
#pragma unroll
      for (int i = 0; i < 8; ++i) {
        v2bf p; p[0] = va[i]; p[1] = vb[i];
        *(v2bf*)(&sVT[(c8 + i) * 32 + r2]) = p;
      }
    }
#else
    for (int t = tid; t < 256; t += 128) {
      const int r2 = (t >> 4) << 1;
      const int c8 = (t & 15) << 3;
      v8bf va = *(const v8bf*)(kvb + (size_t)(keybase + r2) * CC + c8);
      v8bf vb = *(const v8bf*)(kvb + (size_t)(keybase + r2 + 1) * CC + c8);
      *(v8bf*)(&sK[r2 * 128 + c8]) = va;
      *(v8bf*)(&sK[(r2 + 1) * 128 + c8]) = vb;
#pragma unroll
      for (int i = 0; i < 8; ++i) {
        v2bf p; p[0] = va[i]; p[1] = vb[i];
        *(v2bf*)(&sVT[(c8 + i) * 32 + r2]) = p;
      }
    }
#endif
    if (kt + 1 < NIT)
      __builtin_prefetch(kvb + (size_t)(keybase + 32 + (tid >> 2)) * CC + (tid & 3) * 32, 0, 0);
    __syncthreads();

    // S^T tiles: S0T = keys 0-15, S1T = keys 16-31; A = K rows (A-layout), B = Q^T
    v8f s0 = {0.f, 0.f, 0.f, 0.f, 0.f, 0.f, 0.f, 0.f};
    v8f s1 = {0.f, 0.f, 0.f, 0.f, 0.f, 0.f, 0.f, 0.f};
#pragma unroll
    for (int b = 0; b < 4; ++b) {
      const __bf16* kr0 = &sK[m16 * 128 + b * 32];
      v8bf lo0 = *(const v8bf*)(kr0 + khalf * 16);
      v8bf hi0 = *(const v8bf*)(kr0 + 8 + khalf * 16);
      const __bf16* kr1 = &sK[(16 + m16) * 128 + b * 32];
      v8bf lo1 = *(const v8bf*)(kr1 + khalf * 16);
      v8bf hi1 = *(const v8bf*)(kr1 + 8 + khalf * 16);
      v16bf a0, a1;
#pragma unroll
      for (int i = 0; i < 8; ++i) {
        a0[i] = lo0[i]; a0[8 + i] = hi0[i];
        a1[i] = lo1[i]; a1[8 + i] = hi1[i];
      }
      s0 = __builtin_amdgcn_wmma_f32_16x16x32_bf16(false, a0, false, bQ[b], (short)0, s0, false, false);
      s1 = __builtin_amdgcn_wmma_f32_16x16x32_bf16(false, a1, false, bQ[b], (short)0, s1, false, false);
    }

    // online softmax: this lane holds 16 of query m16's 32 key scores
    float x0[8], x1[8];
    float pmax = -3.0e38f;
#pragma unroll
    for (int v = 0; v < 8; ++v) {
      x0[v] = s0[v] * scale;
      x1[v] = s1[v] * scale;
      pmax = fmaxf(pmax, fmaxf(x0[v], x1[v]));
    }
    pmax = fmaxf(pmax, __shfl_xor(pmax, 16));
    const float mnew = fmaxf(mstate, pmax);
    const float corr = __expf(mstate - mnew);
    mstate = mnew;
    float ps = 0.f;
    v8bf e0, e1;
#pragma unroll
    for (int v = 0; v < 8; ++v) {
      const float q0 = __expf(x0[v] - mnew);
      const float q1 = __expf(x1[v] - mnew);
      ps += q0 + q1;
      e0[v] = (__bf16)q0;
      e1[v] = (__bf16)q1;
    }
    ps += __shfl_xor(ps, 16);
    lstate = lstate * corr + ps;
#pragma unroll
    for (int f = 0; f < 8; ++f)
#pragma unroll
      for (int v = 0; v < 8; ++v) oc[f][v] *= corr;

    // store P^T: lane's 8 values are consecutive keys for query m16 -> packed stores
    __bf16* pr = &sP[wave][m16 * 32];
    *(v8bf*)(pr + khalf * 8) = e0;
    *(v8bf*)(pr + 16 + khalf * 8) = e1;

    v16bf bP = *(const v16bf*)(&sP[wave][m16 * 32 + khalf * 16]);

    // O^T += V^T @ P^T ; A = V^T chunk (16 fcols x 32 keys) from sVT
#pragma unroll
    for (int f = 0; f < 8; ++f) {
      const __bf16* vr = &sVT[(f * 16 + m16) * 32];
      v8bf lo = *(const v8bf*)(vr + khalf * 16);
      v8bf hi = *(const v8bf*)(vr + 8 + khalf * 16);
      v16bf aV;
#pragma unroll
      for (int i = 0; i < 8; ++i) { aV[i] = lo[i]; aV[8 + i] = hi[i]; }
      oc[f] = __builtin_amdgcn_wmma_f32_16x16x32_bf16(false, aV, false, bP, (short)0, oc[f], false, false);
    }
  }

  // epilogue: write UNnormalized partial O^T -> po[split][query][fcol], plus (m,l)
  float* orow = po + ((size_t)split * NN + qrow0 + m16) * CC;
#pragma unroll
  for (int f = 0; f < 8; ++f) {
    float4 lo4 = {oc[f][0], oc[f][1], oc[f][2], oc[f][3]};
    float4 hi4 = {oc[f][4], oc[f][5], oc[f][6], oc[f][7]};
    *(float4*)(orow + f * 16 + khalf * 8) = lo4;
    *(float4*)(orow + f * 16 + khalf * 8 + 4) = hi4;
  }
  if (khalf == 0) {
    pm[split * NN + qrow0 + m16] = mstate;
    pl[split * NN + qrow0 + m16] = lstate;
  }
}

// combine key-splits + att_bias + residual + LayerNorm, one row per 128-thread block
__global__ void combine_ln(const float* __restrict__ po, const float* __restrict__ pm,
                           const float* __restrict__ pl, const float* __restrict__ x,
                           const float* __restrict__ bias, const float* __restrict__ gamma,
                           const float* __restrict__ beta, float* __restrict__ out) {
  __shared__ float red[CC];
  const int row = blockIdx.x, c = threadIdx.x;
  float m = -3.0e38f;
#pragma unroll
  for (int s = 0; s < SPLITS; ++s) m = fmaxf(m, pm[s * NN + row]);
  float l = 0.f, o = 0.f;
#pragma unroll
  for (int s = 0; s < SPLITS; ++s) {
    const float wgt = __expf(pm[s * NN + row] - m);
    l += pl[s * NN + row] * wgt;
    o += po[((size_t)s * NN + row) * CC + c] * wgt;
  }
  const float h = o / l + bias[c] + x[(size_t)row * CC + c];
  red[c] = h;
  __syncthreads();
  for (int s = 64; s > 0; s >>= 1) { if (c < s) red[c] += red[c + s]; __syncthreads(); }
  const float mu = red[0] / CC;
  __syncthreads();
  const float dv = h - mu;
  red[c] = dv * dv;
  __syncthreads();
  for (int s = 64; s > 0; s >>= 1) { if (c < s) red[c] += red[c + s]; __syncthreads(); }
  const float var = red[0] / CC;
  const float r = rsqrtf(var + 1e-5f);
  out[(size_t)row * CC + c] = dv * r * gamma[c] + beta[c];
}

extern "C" void kernel_launch(void* const* d_in, const int* in_sizes, int n_in,
                              void* d_out, int out_size, void* d_ws, size_t ws_size,
                              hipStream_t stream) {
  const float* x   = (const float*)d_in[0];
  const int*   ei  = (const int*)d_in[1];
  // d_in[2] = edge_attr (unused, matches reference)
  const float* W1  = (const float*)d_in[3];
  const float* as1 = (const float*)d_in[4];
  const float* ad1 = (const float*)d_in[5];
  const float* b1  = (const float*)d_in[6];
  const float* W2  = (const float*)d_in[7];
  const float* as2 = (const float*)d_in[8];
  const float* ad2 = (const float*)d_in[9];
  const float* b2  = (const float*)d_in[10];
  const float* Wa  = (const float*)d_in[11];
  const float* ba  = (const float*)d_in[12];
  const float* gam = (const float*)d_in[13];
  const float* bet = (const float*)d_in[14];
  float* out = (float*)d_out;

  char* w = (char*)d_ws;
  float* bufA = (float*)w;     w += sizeof(float) * NN * CC;       // xw / q
  float* bufB = (float*)w;     w += sizeof(float) * NN * CC;       // h1 / h2
  float* asN  = (float*)w;     w += sizeof(float) * NN;
  float* adN  = (float*)w;     w += sizeof(float) * NN;
  unsigned* em = (unsigned*)w; w += sizeof(unsigned) * NN;
  float* dn   = (float*)w;     w += sizeof(float) * NN;
  float* ev   = (float*)w;     w += sizeof(float) * ET;
  float* pv   = (float*)w;     w += sizeof(float) * ET;
  __bf16* kvb = (__bf16*)w;    w += sizeof(__bf16) * NN * CC;
  __bf16* qb  = (__bf16*)w;    w += sizeof(__bf16) * NN * CC;
  float* po   = (float*)w;     w += sizeof(float) * SPLITS * NN * CC;
  float* pmb  = (float*)w;     w += sizeof(float) * SPLITS * NN;
  float* plb  = (float*)w;     w += sizeof(float) * SPLITS * NN;

  const dim3 b128(128), b256(256);
  const int gG = NN / 16;
  const int gA = NN / 8;
  const int gE = (ET + 255) / 256;
  const int gAgg = (int)(((long long)ET * 32 + 255) / 256);
  const int gNC = (NN * CC) / 256;
  const dim3 gF(NN / 64, SPLITS);

  // ---- GATConv layer 1 ----
  node_gemm_wmma<<<gG, b128, 0, stream>>>(x, W1, bufA);
  alpha_kernel<<<gA, b256, 0, stream>>>(bufA, as1, ad1, asN, adN);
  gat_init<<<gNC, b256, 0, stream>>>(em, dn, bufB, b1);
  edge_logits<<<gE, b256, 0, stream>>>(ei, asN, adN, ev, em);
  edge_exp<<<gE, b256, 0, stream>>>(ei, ev, em, pv, dn);
  edge_aggregate<<<gAgg, b256, 0, stream>>>(ei, pv, dn, bufA, bufB);
  relu_k<<<gNC, b256, 0, stream>>>(bufB, NN * CC);

  // ---- GATConv layer 2 ----
  node_gemm_wmma<<<gG, b128, 0, stream>>>(bufB, W2, bufA);
  alpha_kernel<<<gA, b256, 0, stream>>>(bufA, as2, ad2, asN, adN);
  gat_init<<<gNC, b256, 0, stream>>>(em, dn, bufB, b2);
  edge_logits<<<gE, b256, 0, stream>>>(ei, asN, adN, ev, em);
  edge_exp<<<gE, b256, 0, stream>>>(ei, ev, em, pv, dn);
  edge_aggregate<<<gAgg, b256, 0, stream>>>(ei, pv, dn, bufA, bufB);  // h2 -> bufB

  // ---- Temporal attention (flash, bf16 WMMA, fp32 softmax/accum, key-split) ----
  node_gemm_wmma<<<gG, b128, 0, stream>>>(bufB, Wa, bufA);            // q -> bufA
  cast_bf16_k<<<gNC, b256, 0, stream>>>(bufB, kvb, NN * CC);
  cast_bf16_k<<<gNC, b256, 0, stream>>>(bufA, qb, NN * CC);
  flash_attn<<<gF, b128, 0, stream>>>(qb, kvb, po, pmb, plb);

  // ---- split combine + residual + LayerNorm ----
  combine_ln<<<NN, b128, 0, stream>>>(po, pmb, plb, x, ba, gam, bet, out);
}
